// GlobalDynamicAttention_33981781246280
// MI455X (gfx1250) — compile-verified, matrix operand order checked by reading, not run on hardware
//
#include <hip/hip_runtime.h>
#include <cstdint>

#define EPS 1e-5f

typedef unsigned short u16;
typedef __attribute__((ext_vector_type(16))) __bf16 v16bf;
typedef __attribute__((ext_vector_type(8)))  float  v8f;
typedef __attribute__((ext_vector_type(4)))  float  v4f;   // clang ext-vector: valid for nontemporal builtins

// ---------------------------------------------------------------------------
// helpers
// ---------------------------------------------------------------------------
__device__ __forceinline__ u16 f2bf(float x) {
    uint32_t u = __float_as_uint(x);
    uint32_t r = u + 0x7FFFu + ((u >> 16) & 1u);   // round-to-nearest-even
    return (u16)(r >> 16);
}

union FragU { uint4 q[2]; v16bf v; };
union HPack { u16 h[8]; uint4 q; };

// Load a 16x32 bf16 A-fragment (also valid as B-fragment when the source is
// the [N][K] row-major matrix) from row-major memory with leading dim `ld`
// (in elements), starting at K-offset `k`.
// Per CDNA5 ISA 16-bit A layout: lane L -> row (L&15), K-base (L>>4)*8,
// elements 0..7 = K kb..kb+7, elements 8..15 = K kb+16..kb+23.
__device__ __forceinline__ v16bf load_frag(const u16* base, int ld, int k) {
    int lane = threadIdx.x & 31;
    int row  = lane & 15;
    int kb   = (lane >> 4) * 8;
    const u16* p = base + (size_t)row * ld + k + kb;
    FragU f;
    f.q[0] = *(const uint4*)(p);
    f.q[1] = *(const uint4*)(p + 16);
    return f.v;
}

__device__ __forceinline__ float rmax16(float v) {
    v = fmaxf(v, __shfl_xor(v, 1));
    v = fmaxf(v, __shfl_xor(v, 2));
    v = fmaxf(v, __shfl_xor(v, 4));
    v = fmaxf(v, __shfl_xor(v, 8));
    return v;
}
__device__ __forceinline__ float rsum16(float v) {
    v += __shfl_xor(v, 1);
    v += __shfl_xor(v, 2);
    v += __shfl_xor(v, 4);
    v += __shfl_xor(v, 8);
    return v;
}

#define WMMA_BF16(A, B, C) \
    __builtin_amdgcn_wmma_f32_16x16x32_bf16(false, (A), false, (B), (short)0, (C), false, false)

// ---------------------------------------------------------------------------
// Kernel 1: weights -> bf16, BN fold (scale/bias), zero pooling accumulators
// ---------------------------------------------------------------------------
__global__ void k_prep_params(
    const float* __restrict__ qw, const float* __restrict__ kw,
    const float* __restrict__ vw, const float* __restrict__ ow,
    const float* __restrict__ qg, const float* __restrict__ qbb,
    const float* __restrict__ qm, const float* __restrict__ qv,
    const float* __restrict__ kg, const float* __restrict__ kbb,
    const float* __restrict__ km, const float* __restrict__ kv,
    const float* __restrict__ vg, const float* __restrict__ vbb,
    const float* __restrict__ vm, const float* __restrict__ vv,
    const float* __restrict__ og, const float* __restrict__ obb,
    const float* __restrict__ om, const float* __restrict__ ov,
    u16* wqb, u16* wkb, u16* wvb, u16* wob,
    float* scq, float* biq, float* sck, float* bik,
    float* scv, float* biv, float* sco, float* bio,
    float* ksum, float* vsum)
{
    int idx = blockIdx.x * 256 + threadIdx.x;
    if (idx < 131072) {
        wqb[idx] = f2bf(qw[idx]);
        wkb[idx] = f2bf(kw[idx]);
        wvb[idx] = f2bf(vw[idx]);
        wob[idx] = f2bf(ow[idx]);
    }
    if (idx < 256) {
        float s;
        s = qg[idx] * rsqrtf(qv[idx] + EPS); scq[idx] = s; biq[idx] = qbb[idx] - qm[idx] * s;
        s = kg[idx] * rsqrtf(kv[idx] + EPS); sck[idx] = s; bik[idx] = kbb[idx] - km[idx] * s;
        s = vg[idx] * rsqrtf(vv[idx] + EPS); scv[idx] = s; biv[idx] = vbb[idx] - vm[idx] * s;
    }
    if (idx < 512) {
        float s = og[idx] * rsqrtf(ov[idx] + EPS); sco[idx] = s; bio[idx] = obb[idx] - om[idx] * s;
    }
    if (idx < 16384) { ksum[idx] = 0.f; vsum[idx] = 0.f; }
}

// ---------------------------------------------------------------------------
// Kernel 2: x (NCHW f32) -> xb bf16 [pix_patchmajor][512]  (LDS transpose)
// pix = patch*1024 + (h%32)*32 + (w%32), patch = (h/32)*8 + (w/32)
// x is streamed once here -> non-temporal loads.
// ---------------------------------------------------------------------------
__global__ void k_prep_x(const float* __restrict__ x, u16* __restrict__ xb)
{
    __shared__ float tile[32][33];
    int c0 = blockIdx.z * 32;
    int h  = blockIdx.y;
    int w0 = blockIdx.x * 32;
    int tx = threadIdx.x;       // 0..31
    int ty = threadIdx.y;       // 0..7
#pragma unroll
    for (int i = 0; i < 4; ++i) {
        int c = ty + i * 8;
        tile[c][tx] = __builtin_nontemporal_load(
            x + (size_t)(c0 + c) * 65536 + h * 256 + (w0 + tx));
    }
    __syncthreads();
    int prow = h >> 5, prh = h & 31;
#pragma unroll
    for (int i = 0; i < 4; ++i) {
        int wi    = ty + i * 8;
        int wg    = w0 + wi;
        int patch = prow * 8 + (wg >> 5);
        int n     = prh * 32 + (wg & 31);
        size_t pix = (size_t)patch * 1024 + n;
        xb[pix * 512 + c0 + tx] = f2bf(tile[tx][wi]);
    }
}

// ---------------------------------------------------------------------------
// Kernel 3: projection GEMM  Y[pix, o] = X[pix,:] @ W[o,:]^T  (M=65536,N=256,K=512)
// mode 0 (q): BN+ReLU, *outscale, store bf16
// mode 1 (k/v): BN+ReLU, fused patch-mean accumulation (LDS + global atomics)
// ---------------------------------------------------------------------------
__global__ __launch_bounds__(256)
void k_proj_qkv(const u16* __restrict__ xb, const u16* __restrict__ wb,
                const float* __restrict__ sc, const float* __restrict__ bi,
                u16* __restrict__ qout, float* __restrict__ pool_sum,
                int mode, float outscale)
{
    int wv   = threadIdx.x >> 5;
    int lane = threadIdx.x & 31;
    int m0   = blockIdx.x * 128 + wv * 16;
    int nb   = blockIdx.y * 64;

    v8f acc[4] = {};
    const u16* A = xb + (size_t)m0 * 512;
    for (int k = 0; k < 512; k += 32) {
        if (k + 32 < 512) __builtin_prefetch(A + ((lane & 15) * 512) + k + 32, 0, 1);
        v16bf a = load_frag(A, 512, k);
#pragma unroll
        for (int t = 0; t < 4; ++t) {
            v16bf b = load_frag(wb + (size_t)(nb + t * 16) * 512, 512, k);
            acc[t] = WMMA_BF16(a, b, acc[t]);
        }
    }

    int col   = lane & 15;
    int rbase = (lane >> 4) * 8;
    if (mode == 0) {
#pragma unroll
        for (int t = 0; t < 4; ++t) {
            int ch = nb + t * 16 + col;
            float s = sc[ch], o = bi[ch];
#pragma unroll
            for (int r = 0; r < 8; ++r) {
                float y = acc[t][r] * s + o;
                y = y > 0.f ? y : 0.f;
                y *= outscale;
                qout[(size_t)(m0 + rbase + r) * 256 + ch] = f2bf(y);
            }
        }
    } else {
        __shared__ float lpool[64];
        for (int i = threadIdx.x; i < 64; i += 256) lpool[i] = 0.f;
        __syncthreads();
        int patch = blockIdx.x >> 3;    // 8 M-blocks per patch (patch-major pixels)
#pragma unroll
        for (int t = 0; t < 4; ++t) {
            int ch = nb + t * 16 + col;
            float s = sc[ch], o = bi[ch];
            float partial = 0.f;
#pragma unroll
            for (int r = 0; r < 8; ++r) {
                float y = acc[t][r] * s + o;
                partial += (y > 0.f ? y : 0.f);
            }
            partial += __shfl_xor(partial, 16);     // combine 16 rows of the tile
            if (lane < 16) atomicAdd(&lpool[t * 16 + col], partial);
        }
        __syncthreads();
        if (threadIdx.x < 64)
            atomicAdd(&pool_sum[(size_t)patch * 256 + nb + threadIdx.x], lpool[threadIdx.x]);
    }
}

// ---------------------------------------------------------------------------
// Kernel 4: pooled sums -> bf16  kpT[m][c] (K-major for B-frags), vpT[c][m]
// ---------------------------------------------------------------------------
__global__ void k_pool_fin(const float* __restrict__ ksum, const float* __restrict__ vsum,
                           u16* __restrict__ kpT, u16* __restrict__ vpT)
{
    int idx = blockIdx.x * 256 + threadIdx.x;   // < 16384
    int m = idx >> 8, c = idx & 255;
    kpT[idx]        = f2bf(ksum[idx] * (1.f / 1024.f));
    vpT[c * 64 + m] = f2bf(vsum[idx] * (1.f / 1024.f));
}

// ---------------------------------------------------------------------------
// Kernel 5: per-patch attn = q @ kp  (M=1024, N=64, K=256)
// stores attn f32, attn bf16, attn^T bf16 (for the flash stage)
// attnT: 8 consecutive rows per lane -> packed 16B store
// ---------------------------------------------------------------------------
__global__ __launch_bounds__(256)
void k_attn_qk(const u16* __restrict__ qb, const u16* __restrict__ kpT,
               float* __restrict__ attnf, u16* __restrict__ attnb,
               u16* __restrict__ attnT)
{
    int wv    = threadIdx.x >> 5;
    int lane  = threadIdx.x & 31;
    int patch = blockIdx.x >> 3;
    int m0    = (blockIdx.x & 7) * 128 + wv * 16;

    v8f acc[4] = {};
    const u16* A = qb + ((size_t)patch * 1024 + m0) * 256;
    for (int k = 0; k < 256; k += 32) {
        v16bf a = load_frag(A, 256, k);
#pragma unroll
        for (int t = 0; t < 4; ++t) {
            v16bf b = load_frag(kpT + (size_t)(t * 16) * 256, 256, k);
            acc[t] = WMMA_BF16(a, b, acc[t]);
        }
    }
    int col = lane & 15, rbase = (lane >> 4) * 8;
    size_t base = (size_t)patch * 65536;
    int row0 = m0 + rbase;                     // multiple of 8
#pragma unroll
    for (int t = 0; t < 4; ++t) {
        int p = t * 16 + col;
        HPack hp;
#pragma unroll
        for (int r = 0; r < 8; ++r) {
            float v = acc[t][r];
            u16 h   = f2bf(v);
            hp.h[r] = h;
            attnf[base + (size_t)(row0 + r) * 64 + p] = v;
            attnb[base + (size_t)(row0 + r) * 64 + p] = h;
        }
        // attnT[p][row0..row0+7]: contiguous 16 bytes per lane
        *(uint4*)(attnT + base + (size_t)p * 1024 + row0) = hp.q;
    }
}

// ---------------------------------------------------------------------------
// Kernel 6: flash-style nested correlation + final softmax
//   corr_map = softmax_p(attn @ attn^T) @ attn       (per patch, 1024x1024x64)
//   attn2    = softmax_d(attn + corr_map)            (d = 64)
// block = 4 waves, 64 query rows; online softmax over p in chunks of 32.
// P tile is re-shaped C-layout -> A-layout through per-wave LDS staging.
// ---------------------------------------------------------------------------
__global__ __launch_bounds__(128)
void k_corr(const float* __restrict__ attnf, const u16* __restrict__ attnb,
            const u16* __restrict__ attnT, u16* __restrict__ attn2)
{
    __shared__ __align__(16) u16 pbuf[4][16 * 32];
    int wv    = threadIdx.x >> 5;
    int lane  = threadIdx.x & 31;
    int patch = blockIdx.x >> 4;
    int q0    = (blockIdx.x & 15) * 64 + wv * 16;

    const u16* Ab = attnb + (size_t)patch * 65536;
    const u16* At = attnT + (size_t)patch * 65536;

    v16bf qa0 = load_frag(Ab + (size_t)q0 * 64, 64, 0);
    v16bf qa1 = load_frag(Ab + (size_t)q0 * 64, 64, 32);

    float mrun[8], lrun[8];
#pragma unroll
    for (int r = 0; r < 8; ++r) { mrun[r] = -1e30f; lrun[r] = 0.f; }
    v8f o[4] = {};

    int col = lane & 15, rbase = (lane >> 4) * 8;

    for (int pc = 0; pc < 1024; pc += 32) {
        // S tiles: S[n, p] = sum_d attn[n,d] * attn[p,d]
        v8f s0 = {}, s1 = {};
        {
            v16bf b;
            b = load_frag(Ab + (size_t)pc * 64, 64, 0);         s0 = WMMA_BF16(qa0, b, s0);
            b = load_frag(Ab + (size_t)pc * 64, 64, 32);        s0 = WMMA_BF16(qa1, b, s0);
            b = load_frag(Ab + (size_t)(pc + 16) * 64, 64, 0);  s1 = WMMA_BF16(qa0, b, s1);
            b = load_frag(Ab + (size_t)(pc + 16) * 64, 64, 32); s1 = WMMA_BF16(qa1, b, s1);
        }
        // online softmax over p
        float p0[8], p1[8], scl[8];
#pragma unroll
        for (int r = 0; r < 8; ++r) {
            float cm = rmax16(fmaxf(s0[r], s1[r]));
            float mn = fmaxf(mrun[r], cm);
            scl[r] = __expf(mrun[r] - mn);
            p0[r]  = __expf(s0[r] - mn);
            p1[r]  = __expf(s1[r] - mn);
            float cs = rsum16(p0[r] + p1[r]);
            lrun[r] = lrun[r] * scl[r] + cs;
            mrun[r] = mn;
        }
#pragma unroll
        for (int t = 0; t < 4; ++t)
#pragma unroll
            for (int r = 0; r < 8; ++r) o[t][r] *= scl[r];

        // stage P (C-layout regs) into LDS as a row-major 16x32 bf16 tile
        __syncthreads();
#pragma unroll
        for (int r = 0; r < 8; ++r) {
            int row = rbase + r;
            pbuf[wv][row * 32 + col]      = f2bf(p0[r]);
            pbuf[wv][row * 32 + col + 16] = f2bf(p1[r]);
        }
        __syncthreads();
        v16bf pa = load_frag(&pbuf[wv][0], 32, 0);

        // O += P @ V   (V^T rows = attnT, K = p chunk)
#pragma unroll
        for (int t = 0; t < 4; ++t) {
            v16bf vb = load_frag(At + (size_t)(t * 16) * 1024, 1024, pc);
            o[t] = WMMA_BF16(pa, vb, o[t]);
        }
    }

    // epilogue: corr_map = O/l ; attn2 = softmax_d(attn + corr_map)
    const float* Af = attnf + (size_t)patch * 65536;
    float vals[4][8];
#pragma unroll
    for (int t = 0; t < 4; ++t)
#pragma unroll
        for (int r = 0; r < 8; ++r) {
            int row = q0 + rbase + r;
            int d   = t * 16 + col;
            vals[t][r] = o[t][r] / lrun[r] + Af[(size_t)row * 64 + d];
        }
#pragma unroll
    for (int r = 0; r < 8; ++r) {
        float mx = fmaxf(fmaxf(vals[0][r], vals[1][r]), fmaxf(vals[2][r], vals[3][r]));
        mx = rmax16(mx);
        float sum = 0.f;
#pragma unroll
        for (int t = 0; t < 4; ++t) { vals[t][r] = __expf(vals[t][r] - mx); sum += vals[t][r]; }
        sum = rsum16(sum);
        float inv = 1.f / sum;
        int row = q0 + rbase + r;
#pragma unroll
        for (int t = 0; t < 4; ++t)
            attn2[(size_t)patch * 65536 + (size_t)row * 64 + t * 16 + col] = f2bf(vals[t][r] * inv);
    }
}

// ---------------------------------------------------------------------------
// Kernel 7: per-patch out = attn2 @ vp  (M=1024, N=256, K=64) -> bf16
// ---------------------------------------------------------------------------
__global__ __launch_bounds__(256)
void k_attn_v(const u16* __restrict__ attn2, const u16* __restrict__ vpT,
              u16* __restrict__ outb)
{
    int wv    = threadIdx.x >> 5;
    int lane  = threadIdx.x & 31;
    int patch = blockIdx.x >> 3;
    int m0    = (blockIdx.x & 7) * 128 + wv * 16;
    int nb    = blockIdx.y * 64;

    v8f acc[4] = {};
    const u16* A = attn2 + (size_t)patch * 65536 + (size_t)m0 * 64;
    for (int k = 0; k < 64; k += 32) {
        v16bf a = load_frag(A, 64, k);
#pragma unroll
        for (int t = 0; t < 4; ++t) {
            v16bf b = load_frag(vpT + (size_t)(nb + t * 16) * 64, 64, k);
            acc[t] = WMMA_BF16(a, b, acc[t]);
        }
    }
    int col = lane & 15, rbase = (lane >> 4) * 8;
#pragma unroll
    for (int t = 0; t < 4; ++t)
#pragma unroll
        for (int r = 0; r < 8; ++r)
            outb[((size_t)patch * 1024 + m0 + rbase + r) * 256 + nb + t * 16 + col] =
                f2bf(acc[t][r]);
}

// ---------------------------------------------------------------------------
// Kernel 8: output projection (M=65536, N=512, K=256) + BN + ReLU + residual,
// store to NCHW f32. Per lane the 8 accumulator rows are 8 consecutive pixels
// -> two float4 (b128) transactions for both the residual load and the store,
// both non-temporal (streamed once, keep L2 for re-read intermediates).
// ---------------------------------------------------------------------------
__global__ __launch_bounds__(256)
void k_proj_o(const u16* __restrict__ outb, const u16* __restrict__ wob,
              const float* __restrict__ sco, const float* __restrict__ bio,
              const float* __restrict__ x, float* __restrict__ out)
{
    int wv   = threadIdx.x >> 5;
    int lane = threadIdx.x & 31;
    int m0   = blockIdx.x * 128 + wv * 16;
    int nb   = blockIdx.y * 64;

    v8f acc[4] = {};
    const u16* A = outb + (size_t)m0 * 256;
    for (int k = 0; k < 256; k += 32) {
        v16bf a = load_frag(A, 256, k);
#pragma unroll
        for (int t = 0; t < 4; ++t) {
            v16bf b = load_frag(wob + (size_t)(nb + t * 16) * 256, 256, k);
            acc[t] = WMMA_BF16(a, b, acc[t]);
        }
    }
    int col = lane & 15, rbase = (lane >> 4) * 8;

    // pix0 = first of 8 consecutive pixels handled by this lane (8-aligned)
    int pix0  = m0 + rbase;
    int patch = pix0 >> 10, n0 = pix0 & 1023;
    int hh = (patch >> 3) * 32 + (n0 >> 5);
    int ww = (patch & 7) * 32 + (n0 & 31);          // multiple of 8, no 32-wrap over r=0..7
    size_t sbase = (size_t)hh * 256 + ww;

#pragma unroll
    for (int t = 0; t < 4; ++t) {
        int ch = nb + t * 16 + col;
        float s = sco[ch], o = bio[ch];
        size_t idx = (size_t)ch * 65536 + sbase;

        v4f xa  = __builtin_nontemporal_load((const v4f*)(x + idx));
        v4f xb4 = __builtin_nontemporal_load((const v4f*)(x + idx + 4));

        float y[8];
#pragma unroll
        for (int r = 0; r < 8; ++r) {
            float v = acc[t][r] * s + o;
            y[r] = v > 0.f ? v : 0.f;
        }
        v4f o0, o1;
        o0.x = y[0] + xa.x;  o0.y = y[1] + xa.y;  o0.z = y[2] + xa.z;  o0.w = y[3] + xa.w;
        o1.x = y[4] + xb4.x; o1.y = y[5] + xb4.y; o1.z = y[6] + xb4.z; o1.w = y[7] + xb4.w;
        __builtin_nontemporal_store(o0, (v4f*)(out + idx));
        __builtin_nontemporal_store(o1, (v4f*)(out + idx + 4));
    }
}

// ---------------------------------------------------------------------------
// launcher
// ---------------------------------------------------------------------------
extern "C" void kernel_launch(void* const* d_in, const int* in_sizes, int n_in,
                              void* d_out, int out_size, void* d_ws, size_t ws_size,
                              hipStream_t stream)
{
    (void)in_sizes; (void)n_in; (void)out_size; (void)ws_size;
    const float* x  = (const float*)d_in[0];
    const float* qw = (const float*)d_in[1];
    const float* qg = (const float*)d_in[2];
    const float* qb_= (const float*)d_in[3];
    const float* qm = (const float*)d_in[4];
    const float* qv = (const float*)d_in[5];
    const float* kw = (const float*)d_in[6];
    const float* kg = (const float*)d_in[7];
    const float* kb_= (const float*)d_in[8];
    const float* km = (const float*)d_in[9];
    const float* kv = (const float*)d_in[10];
    const float* vw = (const float*)d_in[11];
    const float* vg = (const float*)d_in[12];
    const float* vb_= (const float*)d_in[13];
    const float* vm = (const float*)d_in[14];
    const float* vv = (const float*)d_in[15];
    const float* ow = (const float*)d_in[16];
    const float* og = (const float*)d_in[17];
    const float* ob_= (const float*)d_in[18];
    const float* om = (const float*)d_in[19];
    const float* ov = (const float*)d_in[20];
    float* out = (float*)d_out;

    char* ws = (char*)d_ws;
    size_t off = 0;
    auto alloc = [&](size_t bytes) -> char* {
        char* p = ws + off;
        off += (bytes + 255) & ~(size_t)255;
        return p;
    };

    u16*   xb    = (u16*)  alloc((size_t)65536 * 512 * 2);   // x bf16, patch-major pixels
    u16*   qb    = (u16*)  alloc((size_t)65536 * 256 * 2);   // q projection bf16
    u16*   outb  = (u16*)  alloc((size_t)65536 * 256 * 2);   // attention output bf16
    u16*   wqb   = (u16*)  alloc(131072 * 2);
    u16*   wkb   = (u16*)  alloc(131072 * 2);
    u16*   wvb   = (u16*)  alloc(131072 * 2);
    u16*   wob   = (u16*)  alloc(131072 * 2);
    float* scq   = (float*)alloc(256 * 4);
    float* biq   = (float*)alloc(256 * 4);
    float* sck   = (float*)alloc(256 * 4);
    float* bik   = (float*)alloc(256 * 4);
    float* scv   = (float*)alloc(256 * 4);
    float* biv   = (float*)alloc(256 * 4);
    float* sco   = (float*)alloc(512 * 4);
    float* bio   = (float*)alloc(512 * 4);
    float* ksum  = (float*)alloc(16384 * 4);
    float* vsum  = (float*)alloc(16384 * 4);
    u16*   kpT   = (u16*)  alloc(16384 * 2);                 // [m=64][c=256]
    u16*   vpT   = (u16*)  alloc(16384 * 2);                 // [c=256][m=64]
    float* attnf = (float*)alloc((size_t)64 * 65536 * 4);
    u16*   attnb = (u16*)  alloc((size_t)64 * 65536 * 2);
    u16*   attnT = (u16*)  alloc((size_t)64 * 65536 * 2);
    u16*   attn2 = (u16*)  alloc((size_t)64 * 65536 * 2);

    const float qscale = 0.04419417382415922f;  // 512^-0.5

    k_prep_params<<<512, 256, 0, stream>>>(
        qw, kw, vw, ow,
        qg, qb_, qm, qv, kg, kb_, km, kv, vg, vb_, vm, vv, og, ob_, om, ov,
        wqb, wkb, wvb, wob,
        scq, biq, sck, bik, scv, biv, sco, bio, ksum, vsum);

    k_prep_x<<<dim3(8, 256, 16), dim3(32, 8, 1), 0, stream>>>(x, xb);

    k_proj_qkv<<<dim3(512, 4), 256, 0, stream>>>(xb, wqb, scq, biq, qb, ksum, 0, qscale);
    k_proj_qkv<<<dim3(512, 4), 256, 0, stream>>>(xb, wkb, sck, bik, qb, ksum, 1, 1.0f);
    k_proj_qkv<<<dim3(512, 4), 256, 0, stream>>>(xb, wvb, scv, biv, qb, vsum, 1, 1.0f);

    k_pool_fin<<<64, 256, 0, stream>>>(ksum, vsum, kpT, vpT);

    k_attn_qk<<<512, 256, 0, stream>>>(qb, kpT, attnf, attnb, attnT);

    k_corr<<<1024, 128, 0, stream>>>(attnf, attnb, attnT, attn2);

    k_attn_v<<<dim3(512, 4), 256, 0, stream>>>(attn2, vpT, outb);

    k_proj_o<<<dim3(512, 8), 256, 0, stream>>>(outb, wob, sco, bio, x, out);
}
